// CinemaScalarImage_66116726554884
// MI455X (gfx1250) — compile-verified, matrix operand order checked by reading, not run on hardware
//
#include <hip/hip_runtime.h>

// ---------------------------------------------------------------------------
// Fully-fused Instant-NGP hash encode + 2x SIREN MLP for MI455X (gfx1250).
// All GEMMs run on V_WMMA_F32_16X16X32_F16 (wave32). Weights live in LDS as
// f16 (transposed so B-fragments are contiguous 16B ds_load_b128), activations
// stay in per-wave LDS staging between layers.
// ---------------------------------------------------------------------------

#define TBL_SIZE (1u << 19)          // T
#define TBL_MASK (TBL_SIZE - 1u)

typedef __attribute__((ext_vector_type(16))) _Float16 v16h;
typedef __attribute__((ext_vector_type(8)))  _Float16 v8h;
typedef __attribute__((ext_vector_type(4)))  _Float16 v4h;
typedef __attribute__((ext_vector_type(8)))  float    v8f;

__constant__ float c_res[8] = {16.f, 35.f, 78.f, 172.f, 380.f, 840.f, 1854.f, 4096.f};

struct KParams {
    const float* pts;   // [N][3]
    const float* tab;   // [8][T][4]
    const float* w[20]; // 0..5 s1_w, 6..11 s1_b, 12..15 s2_w, 16..19 s2_b
    float*       out;   // [2N]: scalar (N) then density (N)
    int          n;
};

// LDS layout (halves): s1w0t 2048 | s1 hidden 4*4096 | s1w5t 1024 |
// s2w0t 4096 | s2w1t 4096 | s2w2t 4096 | s2w3 64 | bias 532 f32 | 8*1536 scratch
#define SMEM_W_HALVES   (2048 + 4 * 4096 + 1024 + 3 * 4096 + 64)  // 31808
#define SMEM_BIAS_F32   532
#define SMEM_WAVE_HALVES 1536
#define SMEM_BYTES (SMEM_W_HALVES * 2 + SMEM_BIAS_F32 * 4 + 8 * SMEM_WAVE_HALVES * 2)

// Build an f16 A/B fragment: elements {k..k+7} and {k+16..k+23} of a row,
// as two contiguous 16B LDS loads.
__device__ __forceinline__ v16h frag_from(const _Float16* p) {
    v8h lo = *reinterpret_cast<const v8h*>(p);
    v8h hb = *reinterpret_cast<const v8h*>(p + 16);
    v16h r;
#pragma unroll
    for (int i = 0; i < 8; ++i) { r[i] = lo[i]; r[i + 8] = hb[i]; }
    return r;
}

// 64 -> 64 dense layer with sin(30x) activation, in/out via act[16][64] LDS.
__device__ __forceinline__ void hidden64(const _Float16* Wt, const float* bias,
                                         _Float16* act, int ml, int hi) {
    const _Float16* arow = act + ml * 64 + hi * 8;
    v16h a0 = frag_from(arow);
    v16h a1 = frag_from(arow + 32);
    v8f cs[4];
#pragma unroll
    for (int nt = 0; nt < 4; ++nt) {
        int n = nt * 16 + ml;
        const _Float16* wrow = Wt + n * 64 + hi * 8;
        v16h b0 = frag_from(wrow);
        v16h b1 = frag_from(wrow + 32);
        float bv = bias[n];
        v8f c;
#pragma unroll
        for (int j = 0; j < 8; ++j) c[j] = bv;
        c = __builtin_amdgcn_wmma_f32_16x16x32_f16(false, a0, false, b0, (short)0, c, false, false);
        c = __builtin_amdgcn_wmma_f32_16x16x32_f16(false, a1, false, b1, (short)0, c, false, false);
        cs[nt] = c;
    }
#pragma unroll
    for (int nt = 0; nt < 4; ++nt) {
        int n = nt * 16 + ml;
#pragma unroll
        for (int j = 0; j < 8; ++j)
            act[(j + 8 * hi) * 64 + n] = (_Float16)__sinf(30.f * cs[nt][j]);
    }
}

// Cooperative f32->f16 transposed weight load: src [kvalid<=din][dout] row-major
// -> dst [dout][din] (rows k>=kvalid zero-padded).
__device__ __forceinline__ void cvtT(_Float16* dst, const float* src,
                                     int din, int dout, int kvalid,
                                     int tid, int nthreads) {
    int total = din * dout;
    for (int i = tid; i < total; i += nthreads) {
        int n = i / din, k = i - n * din;
        dst[i] = (k < kvalid) ? (_Float16)src[k * dout + n] : (_Float16)0.f;
    }
}

__device__ __forceinline__ void cp_f(float* dst, const float* src, int n,
                                     int tid, int nthreads) {
    for (int i = tid; i < n; i += nthreads) dst[i] = src[i];
}

__global__ __launch_bounds__(256) void ngp_siren_fused(KParams P) {
    extern __shared__ __align__(16) char smem[];
    _Float16* s1w0t = (_Float16*)smem;          // [64][32]
    _Float16* s1wht = s1w0t + 2048;             // 4 x [64][64]
    _Float16* s1w5t = s1wht + 4 * 4096;         // [16][64]
    _Float16* s2w0t = s1w5t + 1024;             // [64][64] (k>=47 zero)
    _Float16* s2w1t = s2w0t + 4096;             // [64][64]
    _Float16* s2w2t = s2w1t + 4096;             // [64][64]
    _Float16* s2w3h = s2w2t + 4096;             // [64]
    float*    sbias = (float*)(s2w3h + 64);     // 532 f32
    _Float16* wscr  = (_Float16*)(sbias + 532);

    const int tid = threadIdx.x;
    const int nth = blockDim.x;

    // ---- preload weights (f16, transposed) and biases (f32) into LDS ----
    cvtT(s1w0t, P.w[0], 32, 64, 32, tid, nth);
#pragma unroll
    for (int l = 0; l < 4; ++l)
        cvtT(s1wht + l * 4096, P.w[1 + l], 64, 64, 64, tid, nth);
    cvtT(s1w5t, P.w[5], 64, 16, 64, tid, nth);
    cvtT(s2w0t, P.w[12], 64, 64, 47, tid, nth);
    cvtT(s2w1t, P.w[13], 64, 64, 64, tid, nth);
    cvtT(s2w2t, P.w[14], 64, 64, 64, tid, nth);
    for (int i = tid; i < 64; i += nth) s2w3h[i] = (_Float16)P.w[15][i];
    {
        const int boff[10] = {0, 64, 128, 192, 256, 320, 336, 400, 464, 528};
        const int bsz[10]  = {64, 64, 64, 64, 64, 16, 64, 64, 64, 1};
        const int bsrc[10] = {6, 7, 8, 9, 10, 11, 16, 17, 18, 19};
#pragma unroll
        for (int a = 0; a < 10; ++a)
            cp_f(sbias + boff[a], P.w[bsrc[a]], bsz[a], tid, nth);
    }
    __syncthreads();

    const int wid  = tid >> 5;
    const int lane = tid & 31;
    const int ml   = lane & 15;     // A row / B,C,D column within tile
    const int hi   = lane >> 4;     // K-half / M-half selector
    _Float16* enc = wscr + wid * SMEM_WAVE_HALVES;  // [16][32] f16
    _Float16* act = enc + 512;                       // [16][64] f16

    const int tiles  = P.n >> 4;          // 16 points per wave-tile
    const int stride = gridDim.x << 3;    // 8 waves per block

    for (int t = blockIdx.x * 8 + wid; t < tiles; t += stride) {
        const int g0 = t << 4;

        // prefetch next tile's points (global_prefetch_b8)
        {
            int tn = t + stride;
            if (tn < tiles)
                __builtin_prefetch(P.pts + (size_t)((tn << 4) + ml) * 3, 0, 0);
        }

        // ================= hash-grid encode =================
        // lane (ml, hi) handles point ml, levels hi*4 .. hi*4+3
        {
            const int g = g0 + ml;
            const float px = (P.pts[g * 3 + 0] + 1.f) * 0.5f;
            const float py = (P.pts[g * 3 + 1] + 1.f) * 0.5f;
            const float pz = (P.pts[g * 3 + 2] + 1.f) * 0.5f;
#pragma unroll
            for (int ll = 0; ll < 4; ++ll) {
                const int l = hi * 4 + ll;
                const float r = c_res[l];
                const float sx = px * r, sy = py * r, sz = pz * r;
                const float fx = floorf(sx), fy = floorf(sy), fz = floorf(sz);
                const float wx = sx - fx, wy = sy - fy, wz = sz - fz;
                const unsigned ix = (unsigned)(int)fx;
                const unsigned iy = (unsigned)(int)fy;
                const unsigned iz = (unsigned)(int)fz;
                const float4* tb = reinterpret_cast<const float4*>(P.tab) + (size_t)l * TBL_SIZE;
                float a0 = 0.f, a1 = 0.f, a2 = 0.f, a3 = 0.f;
#pragma unroll
                for (int c = 0; c < 8; ++c) {
                    const unsigned ox = (c >> 2) & 1, oy = (c >> 1) & 1, oz = c & 1;
                    const unsigned h = (ix + ox) ^ ((iy + oy) * 2654435761u)
                                                ^ ((iz + oz) * 805459861u);
                    const float4 f = tb[h & TBL_MASK];
                    const float w = (ox ? wx : 1.f - wx) * (oy ? wy : 1.f - wy)
                                  * (oz ? wz : 1.f - wz);
                    a0 += f.x * w; a1 += f.y * w; a2 += f.z * w; a3 += f.w * w;
                }
                v4h e; e[0] = (_Float16)a0; e[1] = (_Float16)a1;
                       e[2] = (_Float16)a2; e[3] = (_Float16)a3;
                *reinterpret_cast<v4h*>(enc + ml * 32 + l * 4) = e;
            }
        }

        // ================= siren1 layer 0: 32 -> 64, sin =================
        {
            v16h a0 = frag_from(enc + ml * 32 + hi * 8);
            v8f cs[4];
#pragma unroll
            for (int nt = 0; nt < 4; ++nt) {
                int n = nt * 16 + ml;
                v16h b = frag_from(s1w0t + n * 32 + hi * 8);
                float bv = sbias[n];
                v8f c;
#pragma unroll
                for (int j = 0; j < 8; ++j) c[j] = bv;
                cs[nt] = __builtin_amdgcn_wmma_f32_16x16x32_f16(
                    false, a0, false, b, (short)0, c, false, false);
            }
#pragma unroll
            for (int nt = 0; nt < 4; ++nt) {
                int n = nt * 16 + ml;
#pragma unroll
                for (int j = 0; j < 8; ++j)
                    act[(j + 8 * hi) * 64 + n] = (_Float16)__sinf(30.f * cs[nt][j]);
            }
        }

        // ================= siren1 hidden layers 1..4 =================
        for (int lyr = 0; lyr < 4; ++lyr)
            hidden64(s1wht + lyr * 4096, sbias + 64 + lyr * 64, act, ml, hi);

        // ================= siren1 output: 64 -> 16 (linear) ==============
        {
            const _Float16* arow = act + ml * 64 + hi * 8;
            v16h a0 = frag_from(arow);
            v16h a1 = frag_from(arow + 32);
            const int n = ml;
            const _Float16* wrow = s1w5t + n * 64 + hi * 8;
            float bv = sbias[320 + n];
            v8f c;
#pragma unroll
            for (int j = 0; j < 8; ++j) c[j] = bv;
            c = __builtin_amdgcn_wmma_f32_16x16x32_f16(false, a0, false, frag_from(wrow),      (short)0, c, false, false);
            c = __builtin_amdgcn_wmma_f32_16x16x32_f16(false, a1, false, frag_from(wrow + 32), (short)0, c, false, false);
            // col 0 -> density (relu); cols 1..15 -> siren2 input cols 0..14
#pragma unroll
            for (int j = 0; j < 8; ++j) {
                const int mm = j + 8 * hi;
                const float v = c[j];
                if (n == 0) P.out[P.n + g0 + mm] = fmaxf(v, 0.f);
                else        act[mm * 64 + (n - 1)] = (_Float16)v;
            }
        }

        // concat: cols 15..46 = enc, cols 47..63 = 0
        {
#pragma unroll
            for (int k = 0; k < 16; ++k)
                act[ml * 64 + 15 + hi * 16 + k] = enc[ml * 32 + hi * 16 + k];
#pragma unroll
            for (int k = 0; k < 17; ++k)
                act[ml * 64 + 47 + k] = (_Float16)0.f;
        }

        // ================= siren2: 3x (64 -> 64, sin) =================
        hidden64(s2w0t, sbias + 336, act, ml, hi);
        hidden64(s2w1t, sbias + 400, act, ml, hi);
        hidden64(s2w2t, sbias + 464, act, ml, hi);

        // ================= siren2 output: 64 -> 1 dot product ============
        if (hi == 0) {
            float acc = sbias[528];
#pragma unroll 8
            for (int k = 0; k < 64; ++k)
                acc += (float)act[ml * 64 + k] * (float)s2w3h[k];
            P.out[g0 + ml] = acc;
        }
    }
}

extern "C" void kernel_launch(void* const* d_in, const int* in_sizes, int n_in,
                              void* d_out, int out_size, void* d_ws, size_t ws_size,
                              hipStream_t stream) {
    (void)out_size; (void)d_ws; (void)ws_size;
    KParams P;
    P.pts = (const float*)d_in[0];
    P.tab = (const float*)d_in[1];
    if (n_in >= 22) {
        // pytree leaves flattened individually
        for (int i = 0; i < 20; ++i) P.w[i] = (const float*)d_in[2 + i];
    } else {
        // each list concatenated into one buffer
        const float* s1w = (const float*)d_in[2];
        const float* s1b = (const float*)d_in[3];
        const float* s2w = (const float*)d_in[4];
        const float* s2b = (const float*)d_in[5];
        const int s1w_off[6] = {0, 2048, 6144, 10240, 14336, 18432};
        const int s1b_off[6] = {0, 64, 128, 192, 256, 320};
        const int s2w_off[4] = {0, 3008, 7104, 11200};
        const int s2b_off[4] = {0, 64, 128, 192};
        for (int i = 0; i < 6; ++i) P.w[i]      = s1w + s1w_off[i];
        for (int i = 0; i < 6; ++i) P.w[6 + i]  = s1b + s1b_off[i];
        for (int i = 0; i < 4; ++i) P.w[12 + i] = s2w + s2w_off[i];
        for (int i = 0; i < 4; ++i) P.w[16 + i] = s2b + s2b_off[i];
    }
    P.out = (float*)d_out;
    P.n = in_sizes[0] / 3;

    dim3 grid(1024), block(256);
    ngp_siren_fused<<<grid, block, SMEM_BYTES, stream>>>(P);
}
